// GVPModel_46677704573712
// MI455X (gfx1250) — matrix operand order repaired and checked by reading, not compiled, not run on hardware
//
#include <hip/hip_runtime.h>
#include <cstdint>
#include <cstddef>

// ---------------- problem constants ----------------
#define NNODE 20000
#define NEDGE 320000
#define NGRAPH 64
#define CHUNK 20000          // edges per conv chunk (16 chunks)

typedef __attribute__((ext_vector_type(16))) __bf16 v16bf;
typedef __attribute__((ext_vector_type(8)))  float  v8f;
typedef __bf16 bf16_t;

// ---------------- device helpers ----------------
__device__ __forceinline__ bf16_t tobf(float f) {
  union { float f; unsigned u; } a; a.f = f;
  unsigned r = (a.u + 0x7FFFu + ((a.u >> 16) & 1u)) >> 16;
  union { unsigned short u; bf16_t b; } o; o.u = (unsigned short)r;
  return o.b;
}
__device__ __forceinline__ float sigm(float x) { return 1.f / (1.f + __expf(-x)); }

// ---------------- weight conversion: f32 [K][N] -> bf16 [N][Kp] (zero pad) ----
__global__ void k_convert(const float* __restrict__ W, bf16_t* __restrict__ Wt,
                          int K, int N, int Kp) {
  int idx = blockIdx.x * blockDim.x + threadIdx.x;
  if (idx >= N * Kp) return;
  int n = idx / Kp, k = idx - n * Kp;
  float v = (k < K) ? W[(size_t)k * N + n] : 0.f;
  Wt[(size_t)n * Kp + k] = tobf(v);
}

// ---------------- bf16 WMMA GEMM: Y[M][Nn] = X[M][Kp] * Wt^T + bias ---------
// X row-major bf16 (Kp mult of 32), Wt N-major bf16 [Nn][Kp], acc f32.
// block = 128 threads (4 waves). Block tile = 32 rows x 128 cols.
// Each wave: 2 col-tiles x 2 row-tiles = 4 independent accumulators
// (4 independent WMMA chains per K-step -> hides XDL latency; A/B frags
//  each reused by 2 WMMAs). grid = (ceil(M/32), Nn/128).
union AFrag { uint4 u[2]; v16bf v; };
union AccU  { v8f v; float f[8]; };

__global__ __launch_bounds__(128)
void k_gemm(const bf16_t* __restrict__ X, const bf16_t* __restrict__ Wt,
            const float* __restrict__ bias, float* __restrict__ Y,
            int M, int Kp, int Nn, int act) {
  extern __shared__ bf16_t lx[];                 // 32 x Kp tile of X
  const int m0    = blockIdx.x * 32;
  const int wvid  = threadIdx.x >> 5;
  const int lane  = threadIdx.x & 31;
  const int nbase = blockIdx.y * 128 + wvid * 32;
  const int rowu4 = Kp >> 3;                     // uint4 per row
  for (int c = threadIdx.x; c < 32 * rowu4; c += blockDim.x) {
    int row = c / rowu4, cc = c - row * rowu4;
    uint4 val = {0u, 0u, 0u, 0u};
    int m = m0 + row;
    if (m < M) val = *(const uint4*)(X + (size_t)m * Kp + cc * 8);
    *(uint4*)(lx + row * Kp + cc * 8) = val;
  }
  __syncthreads();

  AccU acc00 = {}, acc01 = {}, acc10 = {}, acc11 = {};
  const int arow = lane & 15;
  const int asel = (lane >> 4) << 4;             // 0 / 16 (A K-sub-offset)
  const int bk   = (lane >> 4) << 4;             // 0 / 16 (B K-row offset)
  const int col0 = nbase + (lane & 15);
  const int col1 = col0 + 16;
  const bf16_t* wp0 = Wt + (size_t)col0 * Kp;
  const bf16_t* wp1 = Wt + (size_t)col1 * Kp;
  for (int kc = 0; kc < Kp; kc += 32) {
    AFrag a0, a1, b0, b1;
    const bf16_t* ap0 = lx + arow * Kp + kc + asel;
    const bf16_t* ap1 = ap0 + 16 * Kp;
    a0.u[0] = *(const uint4*)(ap0);              // K = kc+asel .. +7
    a0.u[1] = *(const uint4*)(ap0 + 8);          // K = kc+asel+8 .. +15
    a1.u[0] = *(const uint4*)(ap1);
    a1.u[1] = *(const uint4*)(ap1 + 8);
    const bf16_t* bp0 = wp0 + kc + bk;
    const bf16_t* bp1 = wp1 + kc + bk;
    b0.u[0] = *(const uint4*)(bp0);              // K = kc+bk .. +7
    b0.u[1] = *(const uint4*)(bp0 + 8);          // K = kc+bk+8 .. +15
    b1.u[0] = *(const uint4*)(bp1);
    b1.u[1] = *(const uint4*)(bp1 + 8);
    if (kc + 32 < Kp) {                          // gfx1250 global_prefetch path
      __builtin_prefetch(bp0 + 32, 0, 1);
      __builtin_prefetch(bp1 + 32, 0, 1);
    }
    acc00.v = __builtin_amdgcn_wmma_f32_16x16x32_bf16(
        false, a0.v, false, b0.v, (short)0, acc00.v, false, false);
    acc01.v = __builtin_amdgcn_wmma_f32_16x16x32_bf16(
        false, a0.v, false, b1.v, (short)0, acc01.v, false, false);
    acc10.v = __builtin_amdgcn_wmma_f32_16x16x32_bf16(
        false, a1.v, false, b0.v, (short)0, acc10.v, false, false);
    acc11.v = __builtin_amdgcn_wmma_f32_16x16x32_bf16(
        false, a1.v, false, b1.v, (short)0, acc11.v, false, false);
  }
  const float bv0 = bias ? bias[col0] : 0.f;
  const float bv1 = bias ? bias[col1] : 0.f;
  const int mb = m0 + ((lane >> 4) << 3);
#pragma unroll
  for (int r = 0; r < 8; ++r) {
    int ma = mb + r;        // rows 0..15 half
    int mc = ma + 16;       // rows 16..31 half
    if (ma < M) {
      float v0 = acc00.f[r] + bv0;
      float v1 = acc01.f[r] + bv1;
      if (act) { v0 = fmaxf(v0, 0.f); v1 = fmaxf(v1, 0.f); }
      Y[(size_t)ma * Nn + col0] = v0;
      Y[(size_t)ma * Nn + col1] = v1;
    }
    if (mc < M) {
      float v0 = acc10.f[r] + bv0;
      float v1 = acc11.f[r] + bv1;
      if (act) { v0 = fmaxf(v0, 0.f); v1 = fmaxf(v1, 0.f); }
      Y[(size_t)mc * Nn + col0] = v0;
      Y[(size_t)mc * Nn + col1] = v1;
    }
  }
}

// ---------------- node embedding: LN + GVP(4,2 -> 128,16), no acts ----------
__global__ void k_node_embed(const float* __restrict__ hVs, const float* __restrict__ hVv,
                             const float* lng, const float* lnb,
                             const float* wh, const float* ws, const float* bs,
                             const float* wv, float* __restrict__ S, float* __restrict__ V) {
  int i = blockIdx.x, t = threadIdx.x;
  __shared__ float sl[4], vl[6], vh[48], vn[16];
  if (t == 0) {
    float a[4]; float mu = 0.f;
#pragma unroll
    for (int k = 0; k < 4; ++k) { a[k] = hVs[(size_t)i * 4 + k]; mu += a[k]; }
    mu *= 0.25f;
    float var = 0.f;
#pragma unroll
    for (int k = 0; k < 4; ++k) { float d = a[k] - mu; var += d * d; }
    var *= 0.25f;
    float inv = rsqrtf(var + 1e-5f);
#pragma unroll
    for (int k = 0; k < 4; ++k) sl[k] = (a[k] - mu) * inv * lng[k] + lnb[k];
    float vv[6];
#pragma unroll
    for (int j = 0; j < 6; ++j) vv[j] = hVv[(size_t)i * 6 + j];
    float n0 = fmaxf(vv[0]*vv[0] + vv[1]*vv[1] + vv[2]*vv[2], 1e-8f);
    float n1 = fmaxf(vv[3]*vv[3] + vv[4]*vv[4] + vv[5]*vv[5], 1e-8f);
    float r = rsqrtf(0.5f * (n0 + n1));
#pragma unroll
    for (int j = 0; j < 6; ++j) vl[j] = vv[j] * r;
  }
  __syncthreads();
  if (t < 48) {
    int d = t / 16, h = t % 16;
    vh[d * 16 + h] = vl[0 * 3 + d] * wh[0 * 16 + h] + vl[1 * 3 + d] * wh[1 * 16 + h];
  }
  __syncthreads();
  if (t < 16) {
    float a0 = vh[t], a1 = vh[16 + t], a2 = vh[32 + t];
    vn[t] = sqrtf(fmaxf(a0 * a0 + a1 * a1 + a2 * a2, 1e-8f));
  }
  __syncthreads();
  float acc = bs[t];
#pragma unroll
  for (int k = 0; k < 4; ++k) acc += sl[k] * ws[k * 128 + t];
#pragma unroll
  for (int h = 0; h < 16; ++h) acc += vn[h] * ws[(4 + h) * 128 + t];
  S[(size_t)i * 128 + t] = acc;
  if (t < 48) {
    int d = t / 16, o = t % 16;
    float a = 0.f;
#pragma unroll
    for (int h = 0; h < 16; ++h) a += vh[d * 16 + h] * wv[h * 16 + o];
    V[(size_t)i * 48 + o * 3 + d] = a;
  }
}

// ---------------- edge embedding: LN + GVP(32,1 -> 32,4), no acts -----------
__global__ void k_edge_embed(const float* __restrict__ hEs, const float* __restrict__ hEv,
                             const float* lng, const float* lnb,
                             const float* wh, const float* ws, const float* bs,
                             const float* wv, float* __restrict__ ES2, float* __restrict__ EV2,
                             int E) {
  int e = blockIdx.x * blockDim.x + threadIdx.x;
  if (e >= E) return;
  float sl[32]; float mu = 0.f;
#pragma unroll
  for (int k = 0; k < 32; ++k) { sl[k] = hEs[(size_t)e * 32 + k]; mu += sl[k]; }
  mu *= (1.f / 32.f);
  float var = 0.f;
#pragma unroll
  for (int k = 0; k < 32; ++k) { float d = sl[k] - mu; var += d * d; }
  var *= (1.f / 32.f);
  float inv = rsqrtf(var + 1e-5f);
#pragma unroll
  for (int k = 0; k < 32; ++k) sl[k] = (sl[k] - mu) * inv * lng[k] + lnb[k];
  float v0 = hEv[(size_t)e * 3 + 0], v1 = hEv[(size_t)e * 3 + 1], v2 = hEv[(size_t)e * 3 + 2];
  float r = rsqrtf(fmaxf(v0 * v0 + v1 * v1 + v2 * v2, 1e-8f));
  v0 *= r; v1 *= r; v2 *= r;
  float vh[12], vn[4];
#pragma unroll
  for (int h = 0; h < 4; ++h) { vh[h] = v0 * wh[h]; vh[4 + h] = v1 * wh[h]; vh[8 + h] = v2 * wh[h]; }
#pragma unroll
  for (int h = 0; h < 4; ++h)
    vn[h] = sqrtf(fmaxf(vh[h]*vh[h] + vh[4+h]*vh[4+h] + vh[8+h]*vh[8+h], 1e-8f));
  for (int j = 0; j < 32; ++j) {
    float acc = bs[j];
#pragma unroll
    for (int k = 0; k < 32; ++k) acc += sl[k] * ws[k * 32 + j];
#pragma unroll
    for (int h = 0; h < 4; ++h) acc += vn[h] * ws[(32 + h) * 32 + j];
    ES2[(size_t)e * 32 + j] = acc;
  }
#pragma unroll
  for (int o = 0; o < 4; ++o) {
    float a0 = 0.f, a1 = 0.f, a2 = 0.f;
#pragma unroll
    for (int h = 0; h < 4; ++h) {
      float w = wv[h * 4 + o];
      a0 += vh[h] * w; a1 += vh[4 + h] * w; a2 += vh[8 + h] * w;
    }
    EV2[(size_t)e * 12 + o * 3 + 0] = a0;
    EV2[(size_t)e * 12 + o * 3 + 1] = a1;
    EV2[(size_t)e * 12 + o * 3 + 2] = a2;
  }
}

// ---------------- degree count ----------------
__global__ void k_count(const int* __restrict__ ei, float* __restrict__ cnt, int E) {
  int e = blockIdx.x * blockDim.x + threadIdx.x;
  if (e < E) atomicAdd(&cnt[ei[E + e]], 1.f);
}

// ---------------- g0 stage: gather + vh + bf16 X row ------------------------
__global__ void k_g0_stage(const float* __restrict__ S, const float* __restrict__ V,
                           const float* __restrict__ ES2, const float* __restrict__ EV2,
                           const int* __restrict__ ei, const float* __restrict__ wh,
                           bf16_t* __restrict__ X, float* __restrict__ VH,
                           int ce, int cc, int E) {
  int i = blockIdx.x; if (i >= cc) return;
  int e = ce + i;
  int src = ei[e], dst = ei[E + e];
  int t = threadIdx.x;
  __shared__ float mv[108], vh[108], vn[36];
  if (t < 108) {
    int c = t / 3, d = t - c * 3;
    float val;
    if (c < 16)      val = V[(size_t)src * 48 + c * 3 + d];
    else if (c < 20) val = EV2[(size_t)e * 12 + (c - 16) * 3 + d];
    else             val = V[(size_t)dst * 48 + (c - 20) * 3 + d];
    mv[c * 3 + d] = val;
  }
  __syncthreads();
  if (t < 108) {
    int d = t / 36, h = t - d * 36;
    float acc = 0.f;
    for (int c = 0; c < 36; ++c) acc += mv[c * 3 + d] * wh[c * 36 + h];
    vh[d * 36 + h] = acc;
    VH[(size_t)i * 108 + d * 36 + h] = acc;
  }
  __syncthreads();
  if (t < 36) {
    float a0 = vh[t], a1 = vh[36 + t], a2 = vh[72 + t];
    vn[t] = sqrtf(fmaxf(a0 * a0 + a1 * a1 + a2 * a2, 1e-8f));
  }
  __syncthreads();
  bf16_t* xr = X + (size_t)i * 352;
  for (int j = t; j < 352; j += 128) {
    float val;
    if (j < 128)      val = S[(size_t)src * 128 + j];
    else if (j < 160) val = ES2[(size_t)e * 32 + (j - 128)];
    else if (j < 288) val = S[(size_t)dst * 128 + (j - 160)];
    else if (j < 324) val = vn[j - 288];
    else              val = 0.f;
    xr[j] = tobf(val);
  }
}

// ---------------- generic stage: vh + vn + bf16 X row -----------------------
__global__ void k_stage(const float* __restrict__ S, int sdim,
                        const float* __restrict__ V, int vch,
                        const float* __restrict__ wh, int h,
                        bf16_t* __restrict__ X, int Kp,
                        float* __restrict__ VH, int M) {
  int i = blockIdx.x; if (i >= M) return;
  int t = threadIdx.x;
  __shared__ float vh[3 * 36], vn[36];
  int tot = 3 * h;
  for (int idx = t; idx < tot; idx += 128) {
    int d = idx / h, hh = idx - d * h;
    float acc = 0.f;
    for (int c = 0; c < vch; ++c) acc += V[(size_t)i * vch * 3 + c * 3 + d] * wh[c * h + hh];
    vh[idx] = acc;
    VH[(size_t)i * tot + idx] = acc;
  }
  __syncthreads();
  for (int hh = t; hh < h; hh += 128) {
    float a0 = vh[hh], a1 = vh[h + hh], a2 = vh[2 * h + hh];
    vn[hh] = sqrtf(fmaxf(a0 * a0 + a1 * a1 + a2 * a2, 1e-8f));
  }
  __syncthreads();
  bf16_t* xr = X + (size_t)i * Kp;
  for (int j = t; j < Kp; j += 128) {
    float val;
    if (j < sdim)          val = S[(size_t)i * sdim + j];
    else if (j < sdim + h) val = vn[j - sdim];
    else                   val = 0.f;
    xr[j] = tobf(val);
  }
}

// ---------------- v_out = vh @ wv (+ optional sigmoid gate) -----------------
__global__ void k_vout(const float* __restrict__ VH, int h, const float* __restrict__ wv,
                       int vo, int gate, float* __restrict__ VO, int M) {
  int idx = blockIdx.x * blockDim.x + threadIdx.x;
  if (idx >= M * vo) return;
  int i = idx / vo, o = idx - i * vo;
  const float* vhp = VH + (size_t)i * 3 * h;
  float a0 = 0.f, a1 = 0.f, a2 = 0.f;
  for (int hh = 0; hh < h; ++hh) {
    float w = wv[hh * vo + o];
    a0 += vhp[hh] * w; a1 += vhp[h + hh] * w; a2 += vhp[2 * h + hh] * w;
  }
  if (gate) {
    float g = sigm(sqrtf(fmaxf(a0 * a0 + a1 * a1 + a2 * a2, 1e-8f)));
    a0 *= g; a1 *= g; a2 *= g;
  }
  VO[(size_t)i * vo * 3 + o * 3 + 0] = a0;
  VO[(size_t)i * vo * 3 + o * 3 + 1] = a1;
  VO[(size_t)i * vo * 3 + o * 3 + 2] = a2;
}

// ---------------- scatter add of messages to dst ----------------------------
__global__ void k_scatter(const float* __restrict__ S2, const float* __restrict__ V2,
                          const int* __restrict__ ei, float* __restrict__ AGS,
                          float* __restrict__ AGV, int ce, int cc, int E) {
  int i = blockIdx.x; if (i >= cc) return;
  int d = ei[E + ce + i];
  int t = threadIdx.x;
  atomicAdd(&AGS[(size_t)d * 128 + t], S2[(size_t)i * 128 + t]);
  if (t < 48) atomicAdd(&AGV[(size_t)d * 48 + t], V2[(size_t)i * 48 + t]);
}

// ---------------- residual (optional, optionally /cnt) + GVP LayerNorm ------
__global__ void k_residual_ln(float* __restrict__ S, float* __restrict__ V,
                              const float* __restrict__ dS, const float* __restrict__ dV,
                              const float* __restrict__ cnt,
                              const float* __restrict__ g, const float* __restrict__ b,
                              int M) {
  int i = blockIdx.x; if (i >= M) return;
  int t = threadIdx.x;
  __shared__ float red[128];
  __shared__ float stat;
  __shared__ float vbuf[48];
  float c = 1.f;
  if (cnt) c = fmaxf(cnt[i], 1.f);
  float x = S[(size_t)i * 128 + t];
  if (dS) x += dS[(size_t)i * 128 + t] / c;
  red[t] = x; __syncthreads();
  for (int o = 64; o > 0; o >>= 1) { if (t < o) red[t] += red[t + o]; __syncthreads(); }
  float mu = red[0] * (1.f / 128.f); __syncthreads();
  float dx = x - mu;
  red[t] = dx * dx; __syncthreads();
  for (int o = 64; o > 0; o >>= 1) { if (t < o) red[t] += red[t + o]; __syncthreads(); }
  float var = red[0] * (1.f / 128.f); __syncthreads();
  S[(size_t)i * 128 + t] = dx * rsqrtf(var + 1e-5f) * g[t] + b[t];
  if (t < 48) {
    float vv = V[(size_t)i * 48 + t];
    if (dV) vv += dV[(size_t)i * 48 + t] / c;
    vbuf[t] = vv;
  }
  __syncthreads();
  if (t < 16) {
    float a0 = vbuf[t * 3], a1 = vbuf[t * 3 + 1], a2 = vbuf[t * 3 + 2];
    red[t] = fmaxf(a0 * a0 + a1 * a1 + a2 * a2, 1e-8f);
  }
  __syncthreads();
  if (t == 0) {
    float s = 0.f;
    for (int k = 0; k < 16; ++k) s += red[k];
    stat = rsqrtf(s * (1.f / 16.f));
  }
  __syncthreads();
  if (t < 48) V[(size_t)i * 48 + t] = vbuf[t] * stat;
}

// ---------------- pooling + head --------------------------------------------
__global__ void k_pool(const float* __restrict__ OUT, const int* __restrict__ batch,
                       float* __restrict__ POOL, float* __restrict__ GCNT, int M) {
  int i = blockIdx.x; if (i >= M) return;
  int t = threadIdx.x;
  int bidx = batch[i];
  atomicAdd(&POOL[(size_t)bidx * 128 + t], OUT[(size_t)i * 128 + t]);
  if (t == 0) atomicAdd(&GCNT[bidx], 1.f);
}

__global__ void k_final(const float* __restrict__ POOL, const float* __restrict__ GCNT,
                        const float* __restrict__ fw, const float* __restrict__ fb,
                        float* __restrict__ out) {
  int idx = blockIdx.x * blockDim.x + threadIdx.x;
  if (idx >= NGRAPH * 10) return;
  int gI = idx / 10, c = idx - gI * 10;
  float cg = fmaxf(GCNT[gI], 1.f);
  float acc = fb[c];
  for (int k = 0; k < 128; ++k) acc += (POOL[(size_t)gI * 128 + k] / cg) * fw[k * 10 + c];
  out[idx] = sigm(acc);
}

// =====================================================================
extern "C" void kernel_launch(void* const* d_in, const int* in_sizes, int n_in,
                              void* d_out, int out_size, void* d_ws, size_t ws_size,
                              hipStream_t stream) {
  (void)in_sizes; (void)n_in; (void)out_size; (void)ws_size;
  const float* hVs = (const float*)d_in[0];
  const float* hVv = (const float*)d_in[1];
  const int*   ei  = (const int*)d_in[2];
  const float* hEs = (const float*)d_in[3];
  const float* hEv = (const float*)d_in[4];
  const int*   batch = (const int*)d_in[5];
#define FP(i) ((const float*)d_in[(i)])
  const float *WvLng = FP(6),  *WvLnb = FP(7);
  const float *Wvwh = FP(8), *Wvws = FP(9), *Wvbs = FP(10), *Wvwv = FP(11);
  const float *WeLng = FP(12), *WeLnb = FP(13);
  const float *Wewh = FP(14), *Wews = FP(15), *Webs = FP(16), *Wewv = FP(17);
  struct LayerP {
    const float *g0wh,*g0ws,*g0bs,*g0wv, *g1wh,*g1ws,*g1bs,*g1wv,
                *g2wh,*g2ws,*g2bs,*g2wv, *ln1g,*ln1b,
                *f0wh,*f0ws,*f0bs,*f0wv, *f1wh,*f1ws,*f1bs,*f1wv, *ln2g,*ln2b;
  } L[2];
  for (int l = 0; l < 2; ++l) {
    int bI = 18 + 24 * l;
    L[l] = { FP(bI+0),FP(bI+1),FP(bI+2),FP(bI+3), FP(bI+4),FP(bI+5),FP(bI+6),FP(bI+7),
             FP(bI+8),FP(bI+9),FP(bI+10),FP(bI+11), FP(bI+12),FP(bI+13),
             FP(bI+14),FP(bI+15),FP(bI+16),FP(bI+17), FP(bI+18),FP(bI+19),FP(bI+20),FP(bI+21),
             FP(bI+22),FP(bI+23) };
  }
  const float *WoLng = FP(66), *WoLnb = FP(67);
  const float *Wowh = FP(68), *Wows = FP(69), *Wobs = FP(70);
  const float *finw = FP(71), *finb = FP(72);
#undef FP

  // -------- workspace carve --------
  uint8_t* base = (uint8_t*)d_ws;
  size_t off = 0;
  auto alloc = [&](size_t bytes) -> void* {
    void* p = base + off;
    off = (off + bytes + 255) & ~(size_t)255;
    return p;
  };
  const size_t N = NNODE, E = NEDGE, C = CHUNK;
  float* S    = (float*)alloc(N * 128 * 4);
  float* V    = (float*)alloc(N * 48 * 4);
  float* ES2  = (float*)alloc(E * 32 * 4);
  float* EV2  = (float*)alloc(E * 12 * 4);
  float* CNT  = (float*)alloc(N * 4);
  float* AGS  = (float*)alloc(N * 128 * 4);
  float* AGV  = (float*)alloc(N * 48 * 4);
  bf16_t* XG0 = (bf16_t*)alloc(C * 352 * 2);
  float* VH0  = (float*)alloc(C * 108 * 4);
  float* S0   = (float*)alloc(C * 128 * 4);
  float* V0   = (float*)alloc(C * 48 * 4);
  bf16_t* XG1 = (bf16_t*)alloc(C * 160 * 2);
  float* VH1  = (float*)alloc(C * 48 * 4);
  float* S1   = (float*)alloc(C * 128 * 4);
  float* V1   = (float*)alloc(C * 48 * 4);
  bf16_t* XG2 = (bf16_t*)alloc(C * 160 * 2);
  float* VH2  = (float*)alloc(C * 48 * 4);
  float* S2   = (float*)alloc(C * 128 * 4);
  float* V2   = (float*)alloc(C * 48 * 4);
  bf16_t* XF0 = (bf16_t*)alloc(N * 160 * 2);
  float* VHF0 = (float*)alloc(N * 96 * 4);
  float* SF0  = (float*)alloc(N * 512 * 4);
  float* VF0  = (float*)alloc(N * 96 * 4);
  bf16_t* XF1 = (bf16_t*)alloc(N * 544 * 2);
  float* VHF1 = (float*)alloc(N * 96 * 4);
  float* SF1  = (float*)alloc(N * 128 * 4);
  float* VF1  = (float*)alloc(N * 48 * 4);
  bf16_t* XO  = (bf16_t*)alloc(N * 160 * 2);
  float* OUTB = (float*)alloc(N * 128 * 4);
  float* POOL = (float*)alloc(NGRAPH * 128 * 4);
  float* GCNT = (float*)alloc(NGRAPH * 4);
  bf16_t* WTg0[2], *WTg1[2], *WTg2[2], *WTf0[2], *WTf1[2];
  for (int l = 0; l < 2; ++l) {
    WTg0[l] = (bf16_t*)alloc(128 * 352 * 2);
    WTg1[l] = (bf16_t*)alloc(128 * 160 * 2);
    WTg2[l] = (bf16_t*)alloc(128 * 160 * 2);
    WTf0[l] = (bf16_t*)alloc(512 * 160 * 2);
    WTf1[l] = (bf16_t*)alloc(128 * 544 * 2);
  }
  bf16_t* WTwo = (bf16_t*)alloc(128 * 160 * 2);

  auto cw = [&](const float* W, bf16_t* Wt, int K, int Nn, int Kp) {
    int tot = Nn * Kp;
    k_convert<<<dim3((tot + 255) / 256), dim3(256), 0, stream>>>(W, Wt, K, Nn, Kp);
  };
  auto gemm = [&](const bf16_t* X, const bf16_t* Wt, const float* bias, float* Y,
                  int M, int Kp, int Nn, int act) {
    k_gemm<<<dim3((M + 31) / 32, Nn / 128), dim3(128), (size_t)32 * Kp * 2, stream>>>(
        X, Wt, bias, Y, M, Kp, Nn, act);
  };

  // -------- weight conversion --------
  for (int l = 0; l < 2; ++l) {
    cw(L[l].g0ws, WTg0[l], 324, 128, 352);
    cw(L[l].g1ws, WTg1[l], 144, 128, 160);
    cw(L[l].g2ws, WTg2[l], 144, 128, 160);
    cw(L[l].f0ws, WTf0[l], 160, 512, 160);
    cw(L[l].f1ws, WTf1[l], 544, 128, 544);
  }
  cw(Wows, WTwo, 144, 128, 160);

  // -------- embeddings + degree --------
  k_node_embed<<<dim3(NNODE), dim3(128), 0, stream>>>(hVs, hVv, WvLng, WvLnb,
                                                      Wvwh, Wvws, Wvbs, Wvwv, S, V);
  k_edge_embed<<<dim3(NEDGE / 256), dim3(256), 0, stream>>>(hEs, hEv, WeLng, WeLnb,
                                                            Wewh, Wews, Webs, Wewv,
                                                            ES2, EV2, NEDGE);
  hipMemsetAsync(CNT, 0, N * 4, stream);
  k_count<<<dim3(NEDGE / 256), dim3(256), 0, stream>>>(ei, CNT, NEDGE);

  // -------- layers --------
  for (int l = 0; l < 2; ++l) {
    hipMemsetAsync(AGS, 0, N * 128 * 4, stream);
    hipMemsetAsync(AGV, 0, N * 48 * 4, stream);
    for (int ch = 0; ch < NEDGE / CHUNK; ++ch) {
      int ce = ch * CHUNK, cc = CHUNK;
      // g0
      k_g0_stage<<<dim3(cc), dim3(128), 0, stream>>>(S, V, ES2, EV2, ei, L[l].g0wh,
                                                     XG0, VH0, ce, cc, NEDGE);
      gemm(XG0, WTg0[l], L[l].g0bs, S0, cc, 352, 128, 1);
      k_vout<<<dim3((cc * 16 + 255) / 256), dim3(256), 0, stream>>>(VH0, 36, L[l].g0wv,
                                                                    16, 1, V0, cc);
      // g1
      k_stage<<<dim3(cc), dim3(128), 0, stream>>>(S0, 128, V0, 16, L[l].g1wh, 16,
                                                  XG1, 160, VH1, cc);
      gemm(XG1, WTg1[l], L[l].g1bs, S1, cc, 160, 128, 1);
      k_vout<<<dim3((cc * 16 + 255) / 256), dim3(256), 0, stream>>>(VH1, 16, L[l].g1wv,
                                                                    16, 1, V1, cc);
      // g2
      k_stage<<<dim3(cc), dim3(128), 0, stream>>>(S1, 128, V1, 16, L[l].g2wh, 16,
                                                  XG2, 160, VH2, cc);
      gemm(XG2, WTg2[l], L[l].g2bs, S2, cc, 160, 128, 0);
      k_vout<<<dim3((cc * 16 + 255) / 256), dim3(256), 0, stream>>>(VH2, 16, L[l].g2wv,
                                                                    16, 0, V2, cc);
      // aggregate
      k_scatter<<<dim3(cc), dim3(128), 0, stream>>>(S2, V2, ei, AGS, AGV, ce, cc, NEDGE);
    }
    // residual + LN1 (mean aggregation via CNT)
    k_residual_ln<<<dim3(NNODE), dim3(128), 0, stream>>>(S, V, AGS, AGV, CNT,
                                                         L[l].ln1g, L[l].ln1b, NNODE);
    // feed-forward
    k_stage<<<dim3(NNODE), dim3(128), 0, stream>>>(S, 128, V, 16, L[l].f0wh, 32,
                                                   XF0, 160, VHF0, NNODE);
    gemm(XF0, WTf0[l], L[l].f0bs, SF0, NNODE, 160, 512, 1);
    k_vout<<<dim3((NNODE * 32 + 255) / 256), dim3(256), 0, stream>>>(VHF0, 32, L[l].f0wv,
                                                                     32, 1, VF0, NNODE);
    k_stage<<<dim3(NNODE), dim3(128), 0, stream>>>(SF0, 512, VF0, 32, L[l].f1wh, 32,
                                                   XF1, 544, VHF1, NNODE);
    gemm(XF1, WTf1[l], L[l].f1bs, SF1, NNODE, 544, 128, 0);
    k_vout<<<dim3((NNODE * 16 + 255) / 256), dim3(256), 0, stream>>>(VHF1, 32, L[l].f1wv,
                                                                     16, 0, VF1, NNODE);
    k_residual_ln<<<dim3(NNODE), dim3(128), 0, stream>>>(S, V, SF1, VF1, nullptr,
                                                         L[l].ln2g, L[l].ln2b, NNODE);
  }

  // -------- output head --------
  k_residual_ln<<<dim3(NNODE), dim3(128), 0, stream>>>(S, V, nullptr, nullptr, nullptr,
                                                       WoLng, WoLnb, NNODE);
  k_stage<<<dim3(NNODE), dim3(128), 0, stream>>>(S, 128, V, 16, Wowh, 16,
                                                 XO, 160, VHF1, NNODE);
  gemm(XO, WTwo, Wobs, OUTB, NNODE, 160, 128, 1);

  hipMemsetAsync(POOL, 0, NGRAPH * 128 * 4, stream);
  hipMemsetAsync(GCNT, 0, NGRAPH * 4, stream);
  k_pool<<<dim3(NNODE), dim3(128), 0, stream>>>(OUTB, batch, POOL, GCNT, NNODE);
  k_final<<<dim3((NGRAPH * 10 + 255) / 256), dim3(256), 0, stream>>>(POOL, GCNT, finw,
                                                                     finb, (float*)d_out);
}
